// SelectiveSSM_72524817760707
// MI455X (gfx1250) — matrix operand order repaired and checked by reading, not compiled, not run on hardware
//
#include <hip/hip_runtime.h>

typedef __attribute__((ext_vector_type(16))) __bf16 v16bf;
typedef __attribute__((ext_vector_type(8)))  float  v8f;
typedef __attribute__((ext_vector_type(4)))  unsigned int v4u;
typedef __attribute__((ext_vector_type(8)))  unsigned int v8u;

union FragBF {
    uint4 u[2];
    v16bf v;
};

__device__ __forceinline__ unsigned short f2bf(float f) {
    unsigned int u = __float_as_uint(f);
    unsigned int r = (u + 0x7FFFu + ((u >> 16) & 1u)) >> 16;   // RNE
    return (unsigned short)r;
}

__device__ __forceinline__ float silu_f(float v) {
    return v * (1.0f / (1.0f + __expf(-v)));
}

__device__ __forceinline__ float softplus_f(float v) {
    return fmaxf(v, 0.0f) + log1pf(__expf(-fabsf(v)));
}

// ---------------------------------------------------------------------------
// Tensor Data Mover: 2D tile (bf16, data_size=2B) global -> LDS.
// D# built per CDNA5 ISA; groups 2/3 omitted (2D).
// Issue from one wave only; tracked by TENSORcnt.
// ---------------------------------------------------------------------------
__device__ __forceinline__ void tdm_load_2d_bf16(
    unsigned int lds_byte_addr,
    const unsigned short* gptr,   // tile start (element pointer)
    unsigned int tensor_d0,       // valid elements per row from tile start
    unsigned int tensor_d1,       // valid rows from tile start (OOB rows -> 0)
    unsigned int tile_d0,         // tile width  (elements)
    unsigned int tile_d1,         // tile height (rows)
    unsigned int stride0)         // row stride (elements)
{
    unsigned long long ga = (unsigned long long)gptr;
    v4u g0;
    g0[0] = 1u;                                            // count=1, user D#
    g0[1] = lds_byte_addr;                                 // lds_addr
    g0[2] = (unsigned int)(ga & 0xFFFFFFFFu);              // global_addr[31:0]
    g0[3] = (unsigned int)((ga >> 32) & 0x01FFFFFFu)       // global_addr[56:32]
          | (2u << 30);                                    // type = 2 ("image")
    v8u g1;
    g1[0] = (1u << 16);                                    // data_size=1 (2B), mask=0
    g1[1] = (tensor_d0 & 0xFFFFu) << 16;                   // tensor_dim0[15:0]
    g1[2] = ((tensor_d0 >> 16) & 0xFFFFu)                  // tensor_dim0[31:16]
          | ((tensor_d1 & 0xFFFFu) << 16);                 // tensor_dim1[15:0]
    g1[3] = ((tensor_d1 >> 16) & 0xFFFFu)                  // tensor_dim1[31:16]
          | ((tile_d0 & 0xFFFFu) << 16);                   // tile_dim0
    g1[4] = (tile_d1 & 0xFFFFu);                           // tile_dim1 (tile_dim2=0)
    g1[5] = stride0;                                       // dim0_stride[31:0]
    g1[6] = 0u;                                            // dim0_stride[47:32], dim1_stride lo
    g1[7] = 0u;
    asm volatile("tensor_load_to_lds %0, %1" :: "s"(g0), "s"(g1) : "memory");
}

// ---------------------------------------------------------------------------
// WMMA GEMM: C[M,N] = A[M,K] * Bt[N,K]^T  (both operands bf16, row-major as noted)
// Macro tile 128x64, BK=64, 8 waves, each wave 32x32 (2x2 WMMA tiles).
// TDM-staged, double-buffered LDS.
// MODE 0: in-proj  -> out0=silu (ng<2048) [+ outbf bf16 copy], out1=silu (ng>=2048)
// MODE 1: delta    -> out0 = softplus(v + bias[ng])
// MODE 2: plain    -> out0[mg*ldo+ng] = v
// ---------------------------------------------------------------------------
#define TM 128
#define TN 64
#define TK 64

template <int MODE>
__global__ __launch_bounds__(256) void wmma_gemm_tdm_kernel(
    const unsigned short* __restrict__ A,   // [M][K] bf16
    const unsigned short* __restrict__ Bt,  // [N][K] bf16
    float* __restrict__ out0, float* __restrict__ out1,
    unsigned short* __restrict__ outbf,
    const float* __restrict__ bias,
    int M, int N, int K, int ldo)
{
    __shared__ unsigned short lA[2][TM * TK];  // [m][k]
    __shared__ unsigned short lB[2][TN * TK];  // [n][k]

    const int tid    = threadIdx.x;
    const int lane   = tid & 31;
    const int wave   = tid >> 5;          // 0..7 : 4 along M, 2 along N
    const int half   = lane >> 4;
    const int l16    = lane & 15;
    const int tileM  = blockIdx.x * TM;
    const int tileN  = blockIdx.y * TN;
    const int m_base = (wave >> 1) * 32;
    const int n_base = (wave & 1) * 32;

    const int nk = K / TK;
    const unsigned int brows = (N > tileN) ? (unsigned int)(N - tileN) : 0u;

    if (wave == 0) {
        tdm_load_2d_bf16((unsigned int)(uintptr_t)&lA[0][0],
                         A + (size_t)tileM * K, TK, TM, TK, TM, (unsigned int)K);
        tdm_load_2d_bf16((unsigned int)(uintptr_t)&lB[0][0],
                         Bt + (size_t)tileN * K, TK, brows, TK, TN, (unsigned int)K);
    }

    v8f acc00 = {}, acc01 = {}, acc10 = {}, acc11 = {};

    for (int kt = 0; kt < nk; ++kt) {
        const int s = kt & 1;
        if (wave == 0) {
            if (kt + 1 < nk) {
                const int kb = (kt + 1) * TK;
                tdm_load_2d_bf16((unsigned int)(uintptr_t)&lA[s ^ 1][0],
                                 A + (size_t)tileM * K + kb, TK, TM, TK, TM,
                                 (unsigned int)K);
                tdm_load_2d_bf16((unsigned int)(uintptr_t)&lB[s ^ 1][0],
                                 Bt + (size_t)tileN * K + kb, TK, brows, TK, TN,
                                 (unsigned int)K);
                __builtin_amdgcn_s_wait_tensorcnt(2);  // stage kt complete
            } else {
                __builtin_amdgcn_s_wait_tensorcnt(0);
            }
        }
        __syncthreads();

        const unsigned short* pA = &lA[s][0];
        const unsigned short* pB = &lB[s][0];
#pragma unroll
        for (int j = 0; j < 2; ++j) {
            const int koff = j * 32;
            FragBF fa0, fa1, fb0, fb1;
            // A fragment layout (16-bit 16x32): lane m=l16, K split by half
            {
                const unsigned short* r0 = pA + (size_t)(m_base + l16) * TK + koff;
                const unsigned short* r1 = pA + (size_t)(m_base + 16 + l16) * TK + koff;
                fa0.u[0] = *(const uint4*)(r0 + 8 * half);
                fa0.u[1] = *(const uint4*)(r0 + 16 + 8 * half);
                fa1.u[0] = *(const uint4*)(r1 + 8 * half);
                fa1.u[1] = *(const uint4*)(r1 + 16 + 8 * half);
            }
            // B fragment layout: lane n=l16, K at 8*half and 16+8*half (matches A)
            {
                const unsigned short* r0 = pB + (size_t)(n_base + l16) * TK + koff;
                const unsigned short* r1 = pB + (size_t)(n_base + 16 + l16) * TK + koff;
                fb0.u[0] = *(const uint4*)(r0 + 8 * half); fb0.u[1] = *(const uint4*)(r0 + 16 + 8 * half);
                fb1.u[0] = *(const uint4*)(r1 + 8 * half); fb1.u[1] = *(const uint4*)(r1 + 16 + 8 * half);
            }
            acc00 = __builtin_amdgcn_wmma_f32_16x16x32_bf16(false, fa0.v, false, fb0.v, (short)0, acc00, false, false);
            acc01 = __builtin_amdgcn_wmma_f32_16x16x32_bf16(false, fa0.v, false, fb1.v, (short)0, acc01, false, false);
            acc10 = __builtin_amdgcn_wmma_f32_16x16x32_bf16(false, fa1.v, false, fb0.v, (short)0, acc10, false, false);
            acc11 = __builtin_amdgcn_wmma_f32_16x16x32_bf16(false, fa1.v, false, fb1.v, (short)0, acc11, false, false);
        }
        __syncthreads();  // done reading stage s before next TDM overwrites it
    }

    // ---- epilogue: C layout — VGPR r: M = r + 8*half, N = lane&15 ----
#pragma unroll
    for (int mi = 0; mi < 2; ++mi) {
#pragma unroll
        for (int ni = 0; ni < 2; ++ni) {
            const v8f acc = (mi == 0) ? (ni == 0 ? acc00 : acc01)
                                      : (ni == 0 ? acc10 : acc11);
            const int ng = tileN + n_base + ni * 16 + l16;
            if (ng >= N) continue;
#pragma unroll
            for (int r = 0; r < 8; ++r) {
                const int mg = tileM + m_base + mi * 16 + 8 * half + r;
                float v = acc[r];
                if (MODE == 0) {
                    const float sv = silu_f(v);
                    if (ng < 2048) {
                        out0[(size_t)mg * 2048 + ng]  = sv;
                        outbf[(size_t)mg * 2048 + ng] = f2bf(sv);
                    } else {
                        out1[(size_t)mg * 2048 + (ng - 2048)] = sv;
                    }
                } else if (MODE == 1) {
                    out0[(size_t)mg * ldo + ng] = softplus_f(v + bias[ng]);
                } else {
                    out0[(size_t)mg * ldo + ng] = v;
                }
            }
        }
    }
}

// ---------------------------------------------------------------------------
// f32 -> bf16 elementwise (for x)
// ---------------------------------------------------------------------------
__global__ __launch_bounds__(256) void convert_bf16_kernel(
    const float* __restrict__ in, unsigned short* __restrict__ out, int n)
{
    const int i = (blockIdx.x * 256 + threadIdx.x) * 4;
    if (i + 3 < n) {
        const float4 v = *(const float4*)(in + i);
        out[i + 0] = f2bf(v.x);
        out[i + 1] = f2bf(v.y);
        out[i + 2] = f2bf(v.z);
        out[i + 3] = f2bf(v.w);
    }
}

// ---------------------------------------------------------------------------
// W[K][N] f32 -> Wt[N][K] bf16 (32x32 LDS tile transpose)
// ---------------------------------------------------------------------------
__global__ __launch_bounds__(256) void transpose_convert_kernel(
    const float* __restrict__ W, unsigned short* __restrict__ Wt, int K, int N)
{
    __shared__ float tile[32][33];
    const int kt = blockIdx.x * 32, nt = blockIdx.y * 32;
    const int tx = threadIdx.x & 31, ty = threadIdx.x >> 5;  // ty 0..7
#pragma unroll
    for (int j = 0; j < 32; j += 8) {
        const int k = kt + ty + j, n = nt + tx;
        tile[ty + j][tx] = (k < K && n < N) ? W[(size_t)k * N + n] : 0.0f;
    }
    __syncthreads();
#pragma unroll
    for (int j = 0; j < 32; j += 8) {
        const int n = nt + ty + j, k = kt + tx;
        if (n < N && k < K) Wt[(size_t)n * K + k] = f2bf(tile[tx][ty + j]);
    }
}

// ---------------------------------------------------------------------------
// selective scan: one lane per (b, d, n);  g = ((b*DI + d) << 4) | n
// h_{t} = exp(dt*A[d,n])*h_{t-1} + dt*B[t,n]*u ; y = sum_n h*C[t,n]
// z = (y + u*D[d]) * silu(res)   stored as bf16 (A-operand of final GEMM)
// ---------------------------------------------------------------------------
__global__ __launch_bounds__(256) void scan_kernel(
    const float* __restrict__ delta, const float* __restrict__ u,
    const float* __restrict__ sres,
    const float* __restrict__ Bm, const float* __restrict__ Cm,
    const float* __restrict__ A_log, const float* __restrict__ Dp,
    unsigned short* __restrict__ z, int L, int DI)
{
    const int g = blockIdx.x * 256 + threadIdx.x;
    const int n = g & 15;
    const int d = (g >> 4) & (DI - 1);
    const int b = g >> 15;

    const float a_n = -__expf(A_log[d * 16 + n]);
    const float Dd  = Dp[d];

    float h = 0.0f;
    for (int t = 0; t < L; ++t) {
        const size_t row = (size_t)b * L + t;
        const float dt = delta[row * DI + d];
        const float ut = u[row * DI + d];
        const float bn = Bm[row * 16 + n];
        const float cn = Cm[row * 16 + n];

        h = __expf(dt * a_n) * h + (dt * bn) * ut;

        float y = h * cn;
        y += __shfl_xor(y, 1);
        y += __shfl_xor(y, 2);
        y += __shfl_xor(y, 4);
        y += __shfl_xor(y, 8);

        if (n == 0) {
            const float sr = sres[row * DI + d];
            z[row * DI + d] = f2bf((y + ut * Dd) * sr);
        }
    }
}

extern "C" void kernel_launch(void* const* d_in, const int* in_sizes, int n_in,
                              void* d_out, int out_size, void* d_ws, size_t ws_size,
                              hipStream_t stream) {
    (void)in_sizes; (void)n_in; (void)out_size; (void)ws_size;
    constexpr int Bsz = 2, L = 2048, DM = 1024, DS = 16, DI = 2048;
    constexpr int M = Bsz * L;  // 4096

    const float* x       = (const float*)d_in[0];
    const float* W_in    = (const float*)d_in[1];
    const float* W_delta = (const float*)d_in[2];
    const float* b_delta = (const float*)d_in[3];
    const float* W_B     = (const float*)d_in[4];
    const float* W_C     = (const float*)d_in[5];
    const float* A_log   = (const float*)d_in[6];
    const float* D_param = (const float*)d_in[7];
    const float* W_out   = (const float*)d_in[8];
    float* out = (float*)d_out;

    // ---- workspace layout ----
    float* u     = (float*)d_ws;                      // [M][DI] f32
    float* sres  = u     + (size_t)M * DI;            // [M][DI] f32
    float* delta = sres  + (size_t)M * DI;            // [M][DI] f32
    float* Bm    = delta + (size_t)M * DI;            // [M][DS] f32
    float* Cm    = Bm    + (size_t)M * DS;            // [M][DS] f32
    unsigned short* xbf  = (unsigned short*)(Cm + (size_t)M * DS);  // [M][DM]
    unsigned short* ubf  = xbf  + (size_t)M * DM;     // [M][DI]
    unsigned short* zbf  = ubf  + (size_t)M * DI;     // [M][DI]
    unsigned short* Wt_in    = zbf      + (size_t)M * DI;      // [2*DI][DM]
    unsigned short* Wt_delta = Wt_in    + (size_t)2 * DI * DM; // [DI][DI]
    unsigned short* Wt_B     = Wt_delta + (size_t)DI * DI;     // [DS][DI]
    unsigned short* Wt_C     = Wt_B     + (size_t)DS * DI;     // [DS][DI]
    unsigned short* Wt_out   = Wt_C     + (size_t)DS * DI;     // [DM][DI]

    dim3 blk(256);

    // ---- stage 0: precision / layout conversion ----
    convert_bf16_kernel<<<dim3((M * DM) / 1024), blk, 0, stream>>>(x, xbf, M * DM);
    transpose_convert_kernel<<<dim3(DM / 32, (2 * DI) / 32), blk, 0, stream>>>(W_in, Wt_in, DM, 2 * DI);
    transpose_convert_kernel<<<dim3(DI / 32, DI / 32), blk, 0, stream>>>(W_delta, Wt_delta, DI, DI);
    transpose_convert_kernel<<<dim3(DI / 32, 1), blk, 0, stream>>>(W_B, Wt_B, DI, DS);
    transpose_convert_kernel<<<dim3(DI / 32, 1), blk, 0, stream>>>(W_C, Wt_C, DI, DS);
    transpose_convert_kernel<<<dim3(DI / 32, DM / 32), blk, 0, stream>>>(W_out, Wt_out, DI, DM);

    // ---- 1) x @ W_in -> u = silu(left) (f32 + bf16), sres = silu(right) ----
    wmma_gemm_tdm_kernel<0><<<dim3(M / TM, (2 * DI) / TN), blk, 0, stream>>>(
        xbf, Wt_in, u, sres, ubf, nullptr, M, 2 * DI, DM, 0);

    // ---- 2) delta = softplus(u @ W_delta + b_delta) ----
    wmma_gemm_tdm_kernel<1><<<dim3(M / TM, DI / TN), blk, 0, stream>>>(
        ubf, Wt_delta, delta, nullptr, nullptr, b_delta, M, DI, DI, DI);

    // ---- 3) Bm = u @ W_B ; Cm = u @ W_C (N=16, TDM zero-fills OOB rows) ----
    wmma_gemm_tdm_kernel<2><<<dim3(M / TM, 1), blk, 0, stream>>>(
        ubf, Wt_B, Bm, nullptr, nullptr, nullptr, M, DS, DI, DS);
    wmma_gemm_tdm_kernel<2><<<dim3(M / TM, 1), blk, 0, stream>>>(
        ubf, Wt_C, Cm, nullptr, nullptr, nullptr, M, DS, DI, DS);

    // ---- 4) selective scan -> z (bf16) ----
    scan_kernel<<<dim3((Bsz * DI * 16) / 256), blk, 0, stream>>>(
        delta, u, sres, Bm, Cm, A_log, D_param, zbf, L, DI);

    // ---- 5) out = z @ W_out ----
    wmma_gemm_tdm_kernel<2><<<dim3(M / TM, DM / TN), blk, 0, stream>>>(
        zbf, Wt_out, out, nullptr, nullptr, nullptr, M, DM, DI, DM);
}
